// ConvSNN_74440373174791
// MI455X (gfx1250) — matrix operand order repaired and checked by reading, not compiled, not run on hardware
//
#include <hip/hip_runtime.h>

// CDNA5 / gfx1250 WMMA types
typedef __attribute__((ext_vector_type(16))) __bf16 v16bf;
typedef __attribute__((ext_vector_type(8)))  float  v8f;

#define TILE_M 128
#define TILE_N 128
#define TILE_K 32
#define LDS_STRIDE 40     // bf16 elems per row (80B): pads b128 frag reads to 2-cycle banks
#define KTOT 1152         // C_in * 3 * 3
#define CIN  128
#define HWD  56
#define NPIX 3136         // 56*56
#define BATCH 32
#define COUT 256
#define NITER 36          // 9 (kh,kw) phases * 4 channel tiles of 32

union FragU { uint4 u[2]; v16bf v; };

__device__ __forceinline__ unsigned short f2bf_rne(float f) {
    unsigned u = __float_as_uint(f);
    u += 0x7FFFu + ((u >> 16) & 1u);
    return (unsigned short)(u >> 16);
}

// one-time weight reorder: w[m][c*9+r] fp32 -> wt[r][m][c] bf16
__global__ __launch_bounds__(256)
void wt_transform(const float* __restrict__ w, unsigned short* __restrict__ wt) {
    const int i = blockIdx.x * 256 + threadIdx.x;   // 9*256*128 = 294912
    const int c = i & 127;
    const int m = (i >> 7) & 255;
    const int r = i >> 15;
    wt[i] = f2bf_rne(w[m * KTOT + c * 9 + r]);
}

// one-time activation transform: x[b][c][hw] fp32 -> xt[b][hw][c] bf16 (channels-last)
__global__ __launch_bounds__(256)
void x_transform(const float* __restrict__ x, unsigned short* __restrict__ xt) {
    __shared__ unsigned short tile[32][33];         // [hw][c], padded
    const int hw0 = blockIdx.x * 32;                // 3136/32 = 98
    const int c0  = blockIdx.y * 32;                // 128/32  = 4
    const int b   = blockIdx.z;                     // 32
    const int tl  = threadIdx.x & 31;
    const int tr  = threadIdx.x >> 5;               // 0..7
    const float* xb = x + (size_t)b * CIN * NPIX;
    #pragma unroll
    for (int i = 0; i < 4; ++i) {                   // coalesced reads along hw
        const int c = c0 + tr + i * 8;
        tile[tl][tr + i * 8] = f2bf_rne(xb[(size_t)c * NPIX + hw0 + tl]);
    }
    __syncthreads();
    unsigned short* o = xt + (size_t)b * NPIX * CIN;
    #pragma unroll
    for (int i = 0; i < 4; ++i) {                   // coalesced writes along c
        const int hwl = tr + i * 8;
        o[(size_t)(hw0 + hwl) * CIN + c0 + tl] = tile[hwl][tl];
    }
}

__global__ __launch_bounds__(256)
void convsnn_wmma_bf16(const unsigned short* __restrict__ xt,
                       const unsigned short* __restrict__ wt,
                       float* __restrict__ out)
{
    // double-buffered bf16 tiles: 2*(10240+10240)B = 40KB LDS
    __shared__ unsigned short As[2][TILE_M * LDS_STRIDE];  // [m][k]
    __shared__ unsigned short Bs[2][TILE_N * LDS_STRIDE];  // [n][k]

    const int t    = threadIdx.x;
    const int lane = t & 31;
    const int wave = t >> 5;        // 0..7
    const int wm   = wave & 3;      // M quadrant (32 rows)
    const int wn   = wave >> 2;     // N half (64 cols)
    const int mBase = blockIdx.y * TILE_M;
    const int nBase = blockIdx.x * TILE_N;

    // ---- B-stage invariants: this thread owns spatial slot nl, k half kslot ----
    const int nl    = t & 127;
    const int kslot = t >> 7;       // 0/1
    const int ng = nBase + nl;
    const int bb = ng / NPIX;
    const int hw = ng - bb * NPIX;
    const int hh = hw / HWD;
    const int ww = hw - hh * HWD;
    const unsigned short* xtb = xt + (size_t)bb * NPIX * CIN;

    // ---- A-stage invariants: row am, 32B half ah ----
    const int am = t >> 1;          // 0..127
    const int ah = t & 1;

    const int mrow = lane & 15;
    const int half = lane >> 4;

    v8f acc[2][4];
    const v8f vzero = {0.f,0.f,0.f,0.f,0.f,0.f,0.f,0.f};
    #pragma unroll
    for (int a = 0; a < 2; ++a)
        #pragma unroll
        for (int b2 = 0; b2 < 4; ++b2) acc[a][b2] = vzero;

    // issue global loads for K-tile `it` (no waits forced here)
    auto loadA = [&](int it, uint4& a0, uint4& a1) {
        const int r  = it >> 2;
        const int ct = it & 3;
        const unsigned short* src =
            wt + (size_t)(r * COUT + mBase + am) * CIN + ct * 32 + ah * 16;
        a0 = *(const uint4*)(src);
        a1 = *(const uint4*)(src + 8);
    };
    auto loadB = [&](int it, uint4& b0, uint4& b1) {
        const int r  = it >> 2;
        const int ct = it & 3;
        const int kh = (r * 11) >> 5;      // r/3 for r in [0,8]
        const int kw = r - 3 * kh;
        const int ih = hh + kh - 1;
        const int iw = ww + kw - 1;
        b0 = make_uint4(0u, 0u, 0u, 0u);
        b1 = make_uint4(0u, 0u, 0u, 0u);
        if ((unsigned)ih < 56u && (unsigned)iw < 56u) {
            const unsigned short* src =
                xtb + (size_t)(ih * HWD + iw) * CIN + ct * 32 + kslot * 16;
            b0 = *(const uint4*)(src);
            b1 = *(const uint4*)(src + 8);
        }
    };
    // commit staged registers to LDS (this is where loadcnt waits land)
    auto storeTiles = [&](int sbuf, uint4 a0, uint4 a1, uint4 b0, uint4 b1) {
        unsigned short* ad = &As[sbuf][am * LDS_STRIDE + ah * 16];
        *(uint4*)(ad)     = a0;
        *(uint4*)(ad + 8) = a1;
        unsigned short* bd = &Bs[sbuf][nl * LDS_STRIDE + kslot * 16];
        *(uint4*)(bd)     = b0;
        *(uint4*)(bd + 8) = b1;
    };

    auto compute = [&](int sbuf) {
        FragU afr[2], bfr[4];
        #pragma unroll
        for (int a = 0; a < 2; ++a) {
            const unsigned short* src =
                &As[sbuf][(wm * 32 + a * 16 + mrow) * LDS_STRIDE + half * 8];
            afr[a].u[0] = *(const uint4*)(src);
            afr[a].u[1] = *(const uint4*)(src + 16);
        }
        #pragma unroll
        for (int b2 = 0; b2 < 4; ++b2) {
            const unsigned short* src =
                &Bs[sbuf][(wn * 64 + b2 * 16 + mrow) * LDS_STRIDE + half * 8];
            bfr[b2].u[0] = *(const uint4*)(src);
            bfr[b2].u[1] = *(const uint4*)(src + 16);
        }
        #pragma unroll
        for (int a = 0; a < 2; ++a)
            #pragma unroll
            for (int b2 = 0; b2 < 4; ++b2)
                acc[a][b2] = __builtin_amdgcn_wmma_f32_16x16x32_bf16(
                    false, afr[a].v, false, bfr[b2].v,
                    (short)0, acc[a][b2], false, false);
    };

    // prologue: stage tile 0
    {
        uint4 a0, a1, b0, b1;
        loadA(0, a0, a1);
        loadB(0, b0, b1);
        storeTiles(0, a0, a1, b0, b1);
    }
    __syncthreads();

    // pipelined main loop: issue loads(it+1) -> WMMA(it) -> commit(it+1) -> barrier
    #pragma unroll 2
    for (int it = 0; it < NITER; ++it) {
        const int buf = it & 1;
        const bool more = (it + 1 < NITER);
        uint4 a0, a1, b0, b1;
        if (more) {
            loadA(it + 1, a0, a1);
            loadB(it + 1, b0, b1);
        }
        compute(buf);
        if (more) storeTiles(buf ^ 1, a0, a1, b0, b1);
        __syncthreads();
    }

    // ---- epilogue: LIF threshold + store (D layout: N=lane%16, M=j+8*half) ----
    #pragma unroll
    for (int b2 = 0; b2 < 4; ++b2) {
        const int n   = nBase + wn * 64 + b2 * 16 + mrow;
        const int ob  = n / NPIX;
        const int ohw = n - ob * NPIX;
        float* op = out + (size_t)ob * COUT * NPIX + ohw;
        #pragma unroll
        for (int a = 0; a < 2; ++a) {
            const int m0 = mBase + wm * 32 + a * 16 + half * 8;
            #pragma unroll
            for (int j = 0; j < 8; ++j) {
                const float v = acc[a][b2][j];
                op[(size_t)(m0 + j) * NPIX] = (v >= 1.0f) ? 1.0f : 0.0f;
            }
        }
    }
}

extern "C" void kernel_launch(void* const* d_in, const int* in_sizes, int n_in,
                              void* d_out, int out_size, void* d_ws, size_t ws_size,
                              hipStream_t stream) {
    (void)in_sizes; (void)n_in; (void)ws_size; (void)out_size;
    const float* x = (const float*)d_in[0];       // [32,128,56,56] fp32
    const float* w = (const float*)d_in[1];       // [256, 1152]    fp32
    float* out = (float*)d_out;                   // [32,256,56,56] fp32 spikes

    // workspace layout: xt (channels-last bf16, 25.7MB) | wt (bf16, 590KB)
    unsigned short* xt = (unsigned short*)d_ws;
    unsigned short* wt = xt + (size_t)BATCH * NPIX * CIN;

    x_transform<<<dim3(NPIX / 32, CIN / 32, BATCH), 256, 0, stream>>>(x, xt);
    wt_transform<<<dim3(9 * COUT * CIN / 256), 256, 0, stream>>>(w, wt);

    dim3 grid(NPIX * BATCH / TILE_N, COUT / TILE_M);  // (784, 2)
    convsnn_wmma_bf16<<<grid, 256, 0, stream>>>(xt, wt, out);
}
